// SS2D_4406636445787
// MI455X (gfx1250) — matrix operand order repaired and checked by reading, not compiled, not run on hardware
//
#include <hip/hip_runtime.h>
#include <hip/hip_bf16.h>

typedef __attribute__((ext_vector_type(16))) _Float16 v16h;
typedef __attribute__((ext_vector_type(8)))  float    v8f;
typedef __attribute__((ext_vector_type(4)))  float    f32x4;

#define D_MODEL 96
#define D_INNER 192
#define NSTATE  16
#define NSPLIT  2                // scan state-split: 2 blocks x 8 states
#define NST2    (NSTATE / NSPLIT)
#define DT_RANK 6
#define KDIR    4
#define BATCH   8
#define LSEQ    4096
#define MROWS   (BATCH * LSEQ)   // 32768
#define XDBL_C  40               // 38 rounded to 40 (6 dt + 16 B + 16 C, 2 pad)
#define CHUNK   64               // scan steps staged per LDS buffer

// ---------------- workspace layout (bytes) ----------------
#define WS_W1P   0                       // in_proj  B-frags: 24*3*32*32B = 73728
#define WS_WXP   131072                  // x_proj   B-frags: 4*3*6*32*32B = 73728
#define WS_W2P   262144                  // out_proj B-frags: 6*6*32*32B  = 36864
#define WS_XA0   524288                  // xa  ch-major [b][192][4096] f32 : 25165824
#define WS_SZ1   (WS_XA0 + 25165824)     // silu(z) row-major [m][192]      : 25165824
#define WS_XC2   (WS_SZ1 + 25165824)     // xc  ch-major  [b][192][4096]    : 25165824
#define WS_XC3   (WS_XC2 + 25165824)     // xc  row-major [b][l][192]       : 25165824
#define WS_XDBL  (WS_XC3 + 25165824)     // x_dbl [b][k][t][40]             : 20971520
#define WS_YPART (WS_XDBL + 20971520)    // y partial [s][k][b][l][192]     : 201326592
#define WS_YG    (WS_YPART + 201326592)  // LN(y)*silu(z) [m][192]          : 25165824

// ---------------- helpers ----------------
__device__ __forceinline__ v8f wmma_f16(v16h a, v16h b, v8f c) {
  return __builtin_amdgcn_wmma_f32_16x16x32_f16(
      false, a, false, b, (short)0, c, false, false);
}

// Per-lane A-fragment (16 f16) of a 16x32 f16 tile from row-major f32.
__device__ __forceinline__ v16h load_a16(const float* __restrict__ row, int kb) {
  f32x4 a0 = *(const f32x4*)(row + kb);
  f32x4 a1 = *(const f32x4*)(row + kb + 4);
  f32x4 a2 = *(const f32x4*)(row + kb + 16);
  f32x4 a3 = *(const f32x4*)(row + kb + 20);
  v16h r;
#pragma unroll
  for (int j = 0; j < 4; ++j) {
    r[j]      = (_Float16)a0[j];
    r[4 + j]  = (_Float16)a1[j];
    r[8 + j]  = (_Float16)a2[j];
    r[12 + j] = (_Float16)a3[j];
  }
  return r;
}

__device__ __forceinline__ float silu_f(float v) {
  return v * __builtin_amdgcn_rcpf(1.0f + __expf(-v));
}

// direction -> spatial index mapping
__device__ __forceinline__ int sp_map(int kd, int t) {
  switch (kd & 3) {
    case 0:  return t;
    case 1:  return ((t & 63) << 6) | (t >> 6);
    case 2:  return (LSEQ - 1) - t;
    default: { int u = (LSEQ - 1) - t; return ((u & 63) << 6) | (u >> 6); }
  }
}

// ---------------- kernel 1: pack weights to f16 WMMA B-fragments ----------------
__global__ void k_prep(const float* __restrict__ w_in,   // (384,96)
                       const float* __restrict__ w_xp,   // (4,38,192)
                       const float* __restrict__ w_out,  // (96,192)
                       _Float16* __restrict__ w1p,
                       _Float16* __restrict__ wxp,
                       _Float16* __restrict__ w2p) {
  int gid = blockIdx.x * blockDim.x + threadIdx.x;
  if (gid < 2304) {                       // in_proj: 24 nt * 3 ks * 32 lane
    int lane = gid & 31, q = gid >> 5;
    int ks = q % 3, nt = q / 3;
    int n = nt * 16 + (lane & 15);
    int kb = ks * 32 + ((lane & 16) ? 8 : 0);
    _Float16* dst = w1p + gid * 16;
#pragma unroll
    for (int j = 0; j < 16; ++j) dst[j] = (_Float16)w_in[n * 96 + kb + (j & 7) + ((j & 8) << 1)];
  } else if (gid < 4608) {                // x_proj: 4 dir * 3 nt * 6 ks * 32 lane
    int e = gid - 2304;
    int lane = e & 31, q = e >> 5;
    int ks = q % 6, q2 = q / 6;
    int nt = q2 % 3, kd = q2 / 3;
    int n = nt * 16 + (lane & 15);
    int kb = ks * 32 + ((lane & 16) ? 8 : 0);
    _Float16* dst = wxp + e * 16;
#pragma unroll
    for (int j = 0; j < 16; ++j)
      dst[j] = (n < 38) ? (_Float16)w_xp[(kd * 38 + n) * 192 + kb + (j & 7) + ((j & 8) << 1)] : (_Float16)0.0f;
  } else if (gid < 5760) {                // out_proj: 6 nt * 6 ks * 32 lane
    int e = gid - 4608;
    int lane = e & 31, q = e >> 5;
    int ks = q % 6, nt = q / 6;
    int n = nt * 16 + (lane & 15);
    int kb = ks * 32 + ((lane & 16) ? 8 : 0);
    _Float16* dst = w2p + e * 16;
#pragma unroll
    for (int j = 0; j < 16; ++j) dst[j] = (_Float16)w_out[n * 192 + kb + (j & 7) + ((j & 8) << 1)];
  }
}

// ---------------- kernel 2: in_proj GEMM (WMMA), A-frag reused over 3 n-tiles ----
__global__ void k_inproj(const float* __restrict__ x,        // (32768, 96)
                         const _Float16* __restrict__ w1p,
                         float* __restrict__ xa0,            // [b][192][4096]
                         float* __restrict__ sz1) {          // [m][192]
  int lane = threadIdx.x & 31, wv = threadIdx.x >> 5;
  int mtile = blockIdx.x;                  // 0..2047
  int mrow = lane & 15;
  int kb = (lane & 16) ? 8 : 0;
  const float* arow = x + (mtile * 16 + mrow) * 96;
  v8f acc[3];
#pragma unroll
  for (int j = 0; j < 3; ++j)
#pragma unroll
    for (int q = 0; q < 8; ++q) acc[j][q] = 0.0f;
#pragma unroll
  for (int ks = 0; ks < 3; ++ks) {
    v16h a = load_a16(arow + ks * 32, kb);
#pragma unroll
    for (int j = 0; j < 3; ++j) {
      int ntile = wv * 3 + j;
      v16h bf = *(const v16h*)(w1p + ((ntile * 3 + ks) * 32 + lane) * 16);
      acc[j] = wmma_f16(a, bf, acc[j]);
    }
  }
  int hl = lane >> 4;
#pragma unroll
  for (int j = 0; j < 3; ++j) {
    int n = (wv * 3 + j) * 16 + (lane & 15);
#pragma unroll
    for (int r = 0; r < 8; ++r) {
      int m = mtile * 16 + r + 8 * hl;
      float v = acc[j][r];
      if (n < D_INNER) {
        int bb = m >> 12, l = m & 4095;
        xa0[((bb * D_INNER + n) << 12) + l] = v;
      } else {
        sz1[m * D_INNER + (n - D_INNER)] = silu_f(v);
      }
    }
  }
}

// ---------------- kernel 3: depthwise 3x3 conv + SiLU ----------------
__global__ void k_conv(const float* __restrict__ xa0,
                       const float* __restrict__ cw,   // (192,1,3,3)
                       const float* __restrict__ cb,   // (192)
                       float* __restrict__ xc2,        // ch-major
                       float* __restrict__ xc3) {      // row-major
  int gid = blockIdx.x * blockDim.x + threadIdx.x;    // ((b*192+d)<<12)+l
  int l = gid & 4095;
  int q = gid >> 12;
  int d = q % D_INNER, bb = q / D_INNER;
  int h = l >> 6, w = l & 63;
  const float* src = xa0 + ((bb * D_INNER + d) << 12);
  const float* wd  = cw + d * 9;
  float s = 0.0f;
#pragma unroll
  for (int i = 0; i < 3; ++i) {
    int hh = h + i - 1;
    if (hh < 0 || hh > 63) continue;
#pragma unroll
    for (int j = 0; j < 3; ++j) {
      int ww = w + j - 1;
      if (ww < 0 || ww > 63) continue;
      s += wd[i * 3 + j] * src[(hh << 6) + ww];
    }
  }
  s += cb[d];
  float v = silu_f(s);
  xc2[((bb * D_INNER + d) << 12) + l] = v;
  xc3[((bb << 12) + l) * D_INNER + d] = v;
}

// ---------------- kernel 4: x_proj GEMM (WMMA), A-frag reused over 3 n-tiles ----
__global__ void k_xdbl(const float* __restrict__ xc3,
                       const _Float16* __restrict__ wxp,
                       float* __restrict__ xdbl) {     // [b][k][t][40]
  int lane = threadIdx.x & 31, wv = threadIdx.x >> 5;
  int task = blockIdx.x * 8 + wv;                      // < 8192
  int mtl = task & 255;
  int q2 = task >> 8;
  int kd = q2 & 3, bb = q2 >> 2;
  int mrow = lane & 15;
  int kb = (lane & 16) ? 8 : 0;
  int t_a = mtl * 16 + mrow;
  const float* arow = xc3 + (bb * LSEQ + sp_map(kd, t_a)) * D_INNER;
  v8f acc[3];
#pragma unroll
  for (int j = 0; j < 3; ++j)
#pragma unroll
    for (int q = 0; q < 8; ++q) acc[j][q] = 0.0f;
#pragma unroll
  for (int ks = 0; ks < 6; ++ks) {
    v16h a = load_a16(arow + ks * 32, kb);
#pragma unroll
    for (int j = 0; j < 3; ++j) {
      v16h bf = *(const v16h*)(wxp + (((kd * 3 + j) * 6 + ks) * 32 + lane) * 16);
      acc[j] = wmma_f16(a, bf, acc[j]);
    }
  }
  int hl = lane >> 4;
  float* orow = xdbl + (size_t)((bb * 4 + kd) * LSEQ) * XDBL_C;
#pragma unroll
  for (int j = 0; j < 3; ++j) {
    int n = j * 16 + (lane & 15);
    if (n < XDBL_C) {
#pragma unroll
      for (int r = 0; r < 8; ++r) {
        int t = mtl * 16 + r + 8 * hl;
        orow[t * XDBL_C + n] = acc[j][r];
      }
    }
  }
}

// ---------------- kernel 5: selective scan ----------------------------------
// State split 2-way across blocks (8 states/lane). Per-step operand rows AND
// input-u rows staged global->LDS with async b128 DMA, double-buffered.
// LDS float layout (single shared array -> base offset 0):
//   [0, 2*2560)                : x_dbl rows, 2 buffers of CHUNK*40
//   [2*2560, 2*2560 + 2*12288) : u rows,     2 buffers of CHUNK*192
#define ROWS_F(buf) ((buf) * CHUNK * XDBL_C)
#define UBUF_F(buf) (2 * CHUNK * XDBL_C + (buf) * CHUNK * D_INNER)
#define SMEM_FLOATS (2 * CHUNK * XDBL_C + 2 * CHUNK * D_INNER)

__device__ __forceinline__ void async_rows(const float* __restrict__ src, int buf) {
  for (int i = threadIdx.x; i < CHUNK * XDBL_C / 4; i += 192) {   // 640 b128s
    unsigned lds_off = (unsigned)((ROWS_F(buf) + i * 4) * 4);
    const float* g = src + i * 4;
    asm volatile("global_load_async_to_lds_b128 %0, %1, off"
                 :: "v"(lds_off), "v"(g) : "memory");
  }
}

__device__ __forceinline__ void async_u(const float* __restrict__ ubase,
                                        int tc, int kd, int buf) {
  int rq = threadIdx.x / 48;          // 4 rows in flight per iteration
  int qq = threadIdx.x % 48;          // 48 b128s per 192-float row
#pragma unroll 4
  for (int it = 0; it < CHUNK / 4; ++it) {
    int row = rq + 4 * it;
    int sp = sp_map(kd, tc * CHUNK + row);
    const float* g = ubase + (size_t)sp * D_INNER + qq * 4;
    unsigned lds_off = (unsigned)((UBUF_F(buf) + row * D_INNER + qq * 4) * 4);
    asm volatile("global_load_async_to_lds_b128 %0, %1, off"
                 :: "v"(lds_off), "v"(g) : "memory");
  }
}

__global__ void __launch_bounds__(192) k_scan(
    const float* __restrict__ xdbl,
    const float* __restrict__ xc3,
    const float* __restrict__ alogs,   // (768,16)
    const float* __restrict__ dtw_,    // (4,192,6)
    const float* __restrict__ dtb_,    // (4,192)
    const float* __restrict__ ds_,     // (768)
    float* __restrict__ ypart) {       // [s][k][b][l][192]
  __shared__ float smem[SMEM_FLOATS];  // ~116 KB
  int d  = threadIdx.x;                // channel 0..191
  int s  = blockIdx.x & 1;             // state half
  int kd = (blockIdx.x >> 1) & 3;
  int bb = blockIdx.x >> 3;
  int ch = kd * D_INNER + d;
  int nb = s * NST2;

  float Ac[NST2], h[NST2], dtw[DT_RANK];
#pragma unroll
  for (int n = 0; n < NST2; ++n) {
    Ac[n] = -__expf(alogs[ch * NSTATE + nb + n]);
    h[n] = 0.0f;
  }
#pragma unroll
  for (int r = 0; r < DT_RANK; ++r) dtw[r] = dtw_[ch * DT_RANK + r];
  float bias = dtb_[ch];
  float Dv   = (s == 0) ? ds_[ch] : 0.0f;    // skip-term added once

  const float* xdb = xdbl + (size_t)((bb * 4 + kd) * LSEQ) * XDBL_C;
  float* yp = ypart + (size_t)(((s * KDIR + kd) * BATCH + bb) * LSEQ) * D_INNER;
  const float* ubase = xc3 + (size_t)(bb * LSEQ) * D_INNER;

  async_rows(xdb, 0);                        // prefetch chunk 0
  async_u(ubase, 0, kd, 0);
  for (int tc = 0; tc < LSEQ / CHUNK; ++tc) {
    int cur = tc & 1;
    asm volatile("s_wait_asynccnt 0" ::: "memory");
    __syncthreads();                         // chunk tc resident; prev buffer free
    if (tc + 1 < LSEQ / CHUNK) {
      async_rows(xdb + (tc + 1) * CHUNK * XDBL_C, cur ^ 1);
      async_u(ubase, tc + 1, kd, cur ^ 1);
    }
    for (int tt = 0; tt < CHUNK; ++tt) {
      int t  = tc * CHUNK + tt;
      int sp = sp_map(kd, t);
      const float* row = &smem[ROWS_F(cur) + tt * XDBL_C];
      float dt = bias;
#pragma unroll
      for (int r = 0; r < DT_RANK; ++r) dt += dtw[r] * row[r];
      float e = __expf(dt);
      float delta = (dt > 15.0f) ? dt : __logf(1.0f + e);
      float u  = smem[UBUF_F(cur) + tt * D_INNER + d];
      float du = delta * u;
      float y = 0.0f;
#pragma unroll
      for (int n = 0; n < NST2; ++n) {
        float dA = __expf(delta * Ac[n]);
        h[n] = dA * h[n] + du * row[DT_RANK + nb + n];
        y += h[n] * row[DT_RANK + NSTATE + nb + n];
      }
      y += Dv * u;
      yp[sp * D_INNER + d] = y;              // de-permuted, coalesced across d
    }
  }
}

// ---------------- kernel 6: combine 2x4 partials + LayerNorm + gate ----------------
__global__ void k_combine(const float* __restrict__ ypart,
                          const float* __restrict__ sz1,
                          const float* __restrict__ lnw,
                          const float* __restrict__ lnb,
                          float* __restrict__ yg) {
  int lane = threadIdx.x & 31, wv = threadIdx.x >> 5;
  int m = blockIdx.x * 8 + wv;           // < 32768
  int bb = m >> 12, l = m & 4095;
  float vals[6];
  float s = 0.0f;
#pragma unroll
  for (int j = 0; j < 6; ++j) {
    int d = lane + 32 * j;
    float v = 0.0f;
#pragma unroll
    for (int sk = 0; sk < 2 * KDIR; ++sk)
      v += ypart[(size_t)((sk * BATCH + bb) * LSEQ + l) * D_INNER + d];
    vals[j] = v;
    s += v;
  }
#pragma unroll
  for (int off = 16; off >= 1; off >>= 1) s += __shfl_xor(s, off, 32);
  float mu = s * (1.0f / 192.0f);
  float vs = 0.0f;
#pragma unroll
  for (int j = 0; j < 6; ++j) { float c = vals[j] - mu; vs += c * c; }
#pragma unroll
  for (int off = 16; off >= 1; off >>= 1) vs += __shfl_xor(vs, off, 32);
  float inv = __builtin_amdgcn_rsqf(vs * (1.0f / 192.0f) + 1e-5f);
#pragma unroll
  for (int j = 0; j < 6; ++j) {
    int d = lane + 32 * j;
    float yn = (vals[j] - mu) * inv * lnw[d] + lnb[d];
    yg[(size_t)m * D_INNER + d] = yn * sz1[(size_t)m * D_INNER + d];
  }
}

// ---------------- kernel 7: out_proj GEMM (WMMA), A-frag reused over 6 n-tiles ----
__global__ void k_outproj(const float* __restrict__ yg,
                          const _Float16* __restrict__ w2p,
                          float* __restrict__ out) {   // (32768, 96)
  int lane = threadIdx.x & 31, wv = threadIdx.x >> 5;
  int mtile = blockIdx.x * 8 + wv;                    // < 2048
  int mrow = lane & 15;
  int kb = (lane & 16) ? 8 : 0;
  const float* arow = yg + (size_t)(mtile * 16 + mrow) * D_INNER;
  v8f acc[6];
#pragma unroll
  for (int j = 0; j < 6; ++j)
#pragma unroll
    for (int q = 0; q < 8; ++q) acc[j][q] = 0.0f;
#pragma unroll
  for (int ks = 0; ks < 6; ++ks) {
    v16h a = load_a16(arow + ks * 32, kb);
#pragma unroll
    for (int j = 0; j < 6; ++j) {
      v16h bf = *(const v16h*)(w2p + ((j * 6 + ks) * 32 + lane) * 16);
      acc[j] = wmma_f16(a, bf, acc[j]);
    }
  }
  int hl = lane >> 4;
#pragma unroll
  for (int j = 0; j < 6; ++j) {
    int n = j * 16 + (lane & 15);
#pragma unroll
    for (int r = 0; r < 8; ++r) {
      int m = mtile * 16 + r + 8 * hl;
      out[(size_t)m * D_MODEL + n] = acc[j][r];
    }
  }
}

// ---------------- launch ----------------
extern "C" void kernel_launch(void* const* d_in, const int* in_sizes, int n_in,
                              void* d_out, int out_size, void* d_ws, size_t ws_size,
                              hipStream_t stream) {
  const float* x      = (const float*)d_in[0];
  const float* w_in   = (const float*)d_in[1];
  const float* conv_w = (const float*)d_in[2];
  const float* conv_b = (const float*)d_in[3];
  const float* xpw    = (const float*)d_in[4];
  const float* dtw    = (const float*)d_in[5];
  const float* dtb    = (const float*)d_in[6];
  const float* alogs  = (const float*)d_in[7];
  const float* dsv    = (const float*)d_in[8];
  const float* lnw    = (const float*)d_in[9];
  const float* lnb    = (const float*)d_in[10];
  const float* w_out  = (const float*)d_in[11];
  float* out = (float*)d_out;

  char* ws = (char*)d_ws;
  _Float16* w1p = (_Float16*)(ws + WS_W1P);
  _Float16* wxp = (_Float16*)(ws + WS_WXP);
  _Float16* w2p = (_Float16*)(ws + WS_W2P);
  float* xa0   = (float*)(ws + WS_XA0);
  float* sz1   = (float*)(ws + WS_SZ1);
  float* xc2   = (float*)(ws + WS_XC2);
  float* xc3   = (float*)(ws + WS_XC3);
  float* xdbl  = (float*)(ws + WS_XDBL);
  float* ypart = (float*)(ws + WS_YPART);
  float* yg    = (float*)(ws + WS_YG);

  k_prep<<<23, 256, 0, stream>>>(w_in, xpw, w_out, w1p, wxp, w2p);
  k_inproj<<<2048, 256, 0, stream>>>(x, w1p, xa0, sz1);
  k_conv<<<(BATCH * D_INNER * LSEQ) / 256, 256, 0, stream>>>(xa0, conv_w, conv_b, xc2, xc3);
  k_xdbl<<<1024, 256, 0, stream>>>(xc3, wxp, xdbl);
  k_scan<<<BATCH * KDIR * NSPLIT, 192, 0, stream>>>(xdbl, xc3, alogs, dtw, dtb, dsv, ypart);
  k_combine<<<MROWS / 8, 256, 0, stream>>>(ypart, sz1, lnw, lnb, yg);
  k_outproj<<<2048 / 8, 256, 0, stream>>>(yg, w2p, out);
}
